// LSTMModel_10376640987538
// MI455X (gfx1250) — matrix operand order repaired and checked by reading, not compiled, not run on hardware
//
#include <hip/hip_runtime.h>

// ---------------------------------------------------------------------------
// Fused 2-layer LSTM (B=2048, T=512, H=64, I=6) + final 64->6 linear.
// One workgroup = 16 batch rows, 4 waves (128 threads).
// Wave w owns hidden units [16w, 16w+16): it computes the i/f/g/o gate
// columns for those units via v_wmma_f32_16x16x32_f16 with all weight
// B-operands pinned in VGPRs for the whole T=512 scan. Recurrent h state is
// double-buffered in LDS (f16) -> only 2 barriers per timestep; c state lives
// in f32 accumulator-layout registers. Activations use hw v_tanh_f32.
// ---------------------------------------------------------------------------

typedef __attribute__((ext_vector_type(16))) _Float16 v16h;
typedef __attribute__((ext_vector_type(8)))  _Float16 v8h;
typedef __attribute__((ext_vector_type(8)))  float    v8f;

#define TSTEPS 512
#define BATCH  2048
#define HID    64
#define INP    6
#define OUTP   6

#define WMMA_F16(a, b, c) \
  __builtin_amdgcn_wmma_f32_16x16x32_f16(false, (a), false, (b), (short)0, (c), false, false)

__device__ __forceinline__ float fast_tanh(float x) {
#if __has_builtin(__builtin_amdgcn_tanhf)
  return __builtin_amdgcn_tanhf(x);          // gfx1250 v_tanh_f32 (1 TRANS op)
#else
  x = fminf(fmaxf(x, -12.0f), 12.0f);
  float e = __expf(2.0f * x);
  return (e - 1.0f) * __builtin_amdgcn_rcpf(e + 1.0f);   // fast rcp, no IEEE div
#endif
}
__device__ __forceinline__ float fast_sigm(float x) {
  // sigmoid(x) = 0.5*tanh(x/2) + 0.5  -> mul + tanh + fma
  return __builtin_fmaf(0.5f, fast_tanh(0.5f * x), 0.5f);
}

// Load one 32x16 f16 B-operand tile (transposed weight slice) from a row-major
// f32 weight matrix W[256][stride]. Lane l holds column n (caller maps);
// lanes 0-15 carry K = k0..k0+15, lanes 16-31 carry K = k0+16..k0+31
// (caller folds the +16 into k0 via `khalf`). Out-of-range K -> 0 (Wih0 pad).
__device__ __forceinline__ v16h load_w_tile(const float* __restrict__ W, int n,
                                            int k0, int stride, int kvalid) {
  v16h r;
#pragma unroll
  for (int j = 0; j < 16; ++j) {
    int k  = k0 + j;
    int kc = (k < kvalid) ? k : (kvalid - 1);   // clamp: keep loads in bounds
    float w = W[n * stride + kc];
    r[j] = (_Float16)((k < kvalid) ? w : 0.0f);
  }
  return r;
}

// Load a 16x32 f16 A-operand chunk from a row-major LDS tile.
// K order used here (contiguous, as in load_w_tile): lanes 0-15: K=0..15 in V0..7;
// lanes 16-31: K=16..31 in V0..7.  Row-major LDS rows make each lane's
// data two contiguous 16-byte segments -> two ds_load_b128.
__device__ __forceinline__ v16h load_a16x32(const _Float16* p, int lane,
                                            int rowStrideHalves, int kbase) {
  int row = lane & 15;
  int s0  = kbase + ((lane & 16) ? 16 : 0);
  const _Float16* q = p + row * rowStrideHalves + s0;
  v8h lo = *(const v8h*)(q);
  v8h hi = *(const v8h*)(q + 8);
  v16h r;
#pragma unroll
  for (int j = 0; j < 8; ++j) { r[j] = lo[j]; r[j + 8] = hi[j]; }
  return r;
}

__global__ void __launch_bounds__(128)
lstm2_fused_kernel(const float* __restrict__ x,
                   const float* __restrict__ Wih0, const float* __restrict__ Whh0,
                   const float* __restrict__ bih0, const float* __restrict__ bhh0,
                   const float* __restrict__ Wih1, const float* __restrict__ Whh1,
                   const float* __restrict__ bih1, const float* __restrict__ bhh1,
                   const float* __restrict__ Wlin, const float* __restrict__ blin,
                   float* __restrict__ out) {
  __shared__ __align__(16) _Float16 h0buf[2][16 * 64];  // layer-0 h, dbl-buffered
  __shared__ __align__(16) _Float16 h1buf[2][16 * 64];  // layer-1 h, dbl-buffered
  __shared__ __align__(16) _Float16 xbuf[16 * 32];      // x_t zero-padded to K=32
  __shared__ float h2f[16 * 64];                        // final h f32 for linear

  const int tid  = threadIdx.x;
  const int lane = tid & 31;
  const int wave = tid >> 5;            // 0..3 : hidden-unit slice
  const int b0   = blockIdx.x * 16;     // batch tile base

  const int jcol  = wave * 16 + (lane & 15);     // hidden unit 0..63 (= N col)
  const int khalf = (lane & 16) ? 16 : 0;        // K offset for B-operand lanes

  // ---- pin all weight B-operand tiles in VGPRs (one-time, ~224 VGPRs) ----
  v16h wih0[4], whh0[2][4], wih1[2][4], whh1[2][4];
  v8f bias0v[4], bias1v[4];
#pragma unroll
  for (int gi = 0; gi < 4; ++gi) {               // gate order: i, f, g, o
    int n = gi * 64 + jcol;                      // row of W == gate column
    float bb0 = bih0[n] + bhh0[n];
    float bb1 = bih1[n] + bhh1[n];
#pragma unroll
    for (int e = 0; e < 8; ++e) { bias0v[gi][e] = bb0; bias1v[gi][e] = bb1; }
    wih0[gi] = load_w_tile(Wih0, n, khalf, INP, INP);    // K=6 padded to 32
#pragma unroll
    for (int kc = 0; kc < 2; ++kc) {
      int k0 = kc * 32 + khalf;
      whh0[kc][gi] = load_w_tile(Whh0, n, k0, HID, HID);
      wih1[kc][gi] = load_w_tile(Wih1, n, k0, HID, HID);
      whh1[kc][gi] = load_w_tile(Whh1, n, k0, HID, HID);
    }
  }

  // ---- zero initial state ----
  for (int idx = tid; idx < 16 * 64; idx += 128) {
    h0buf[0][idx] = (_Float16)0.0f;  h0buf[1][idx] = (_Float16)0.0f;
    h1buf[0][idx] = (_Float16)0.0f;  h1buf[1][idx] = (_Float16)0.0f;
  }
  for (int idx = tid; idx < 16 * 32; idx += 128) xbuf[idx] = (_Float16)0.0f;
  v8f c0 = {}, c1 = {};
  v8f h1last = {};
  const int mrowbase = (lane & 16) ? 8 : 0;      // C-layout row base for writes
  __syncthreads();

#pragma unroll 1
  for (int t = 0; t < TSTEPS; ++t) {
    const int cur = t & 1, nxt = cur ^ 1;
    // stage x_t (16 rows x 6 f32 -> f16 into padded 16x32 tile)
    if (tid < 96) {
      int r = tid / 6, i = tid - r * 6;
      xbuf[r * 32 + i] =
          (_Float16)x[(size_t)(b0 + r) * (TSTEPS * INP) + t * INP + i];
    }
    if (t + 1 < TSTEPS && tid < 16)   // global_prefetch next timestep's x rows
      __builtin_prefetch(&x[(size_t)(b0 + tid) * (TSTEPS * INP) + (t + 1) * INP], 0, 0);
    __syncthreads();  // B1: x_t, h0buf[cur], h1buf[cur] all visible

    // ============================ layer 0 ============================
    v16h ax  = load_a16x32(xbuf, lane, 32, 0);
    v16h a00 = load_a16x32(h0buf[cur], lane, 64, 0);
    v16h a01 = load_a16x32(h0buf[cur], lane, 64, 32);
    v8f acc[4];
#pragma unroll
    for (int gi = 0; gi < 4; ++gi) {
      acc[gi] = WMMA_F16(ax,  wih0[gi],    bias0v[gi]);  // x_t @ Wih0^T + b
      acc[gi] = WMMA_F16(a00, whh0[0][gi], acc[gi]);     // h0 @ Whh0^T (K 0..31)
      acc[gi] = WMMA_F16(a01, whh0[1][gi], acc[gi]);     // h0 @ Whh0^T (K 32..63)
    }
#pragma unroll
    for (int e = 0; e < 8; ++e) {
      float ig = fast_sigm(acc[0][e]);
      float fg = fast_sigm(acc[1][e]);
      float gg = fast_tanh(acc[2][e]);
      float og = fast_sigm(acc[3][e]);
      float c  = __builtin_fmaf(fg, c0[e], ig * gg);
      c0[e] = c;
      // write new layer-0 h straight to the alternate buffer (no WAR barrier)
      h0buf[nxt][(mrowbase + e) * 64 + jcol] = (_Float16)(og * fast_tanh(c));
    }
    __syncthreads();  // B2: new layer-0 h visible

    // ============================ layer 1 ============================
    v16h ai0 = load_a16x32(h0buf[nxt], lane, 64, 0);   // layer-1 input = h0(t)
    v16h ai1 = load_a16x32(h0buf[nxt], lane, 64, 32);
    v16h a10 = load_a16x32(h1buf[cur], lane, 64, 0);   // recurrent h1(t-1)
    v16h a11 = load_a16x32(h1buf[cur], lane, 64, 32);
#pragma unroll
    for (int gi = 0; gi < 4; ++gi) {
      acc[gi] = WMMA_F16(ai0, wih1[0][gi], bias1v[gi]);
      acc[gi] = WMMA_F16(ai1, wih1[1][gi], acc[gi]);
      acc[gi] = WMMA_F16(a10, whh1[0][gi], acc[gi]);
      acc[gi] = WMMA_F16(a11, whh1[1][gi], acc[gi]);
    }
#pragma unroll
    for (int e = 0; e < 8; ++e) {
      float ig = fast_sigm(acc[0][e]);
      float fg = fast_sigm(acc[1][e]);
      float gg = fast_tanh(acc[2][e]);
      float og = fast_sigm(acc[3][e]);
      float c  = __builtin_fmaf(fg, c1[e], ig * gg);
      c1[e]     = c;
      h1last[e] = og * fast_tanh(c);
      h1buf[nxt][(mrowbase + e) * 64 + jcol] = (_Float16)h1last[e];
    }
    // loop-top B1 orders these writes against next step's reads
  }

  // ---- final linear: out = h2[:, -1, :] @ Wlin^T + blin (f32 path) ----
#pragma unroll
  for (int e = 0; e < 8; ++e)
    h2f[(mrowbase + e) * 64 + jcol] = h1last[e];
  __syncthreads();

  if (tid < 96) {            // 16 rows x 6 outputs
    int r = tid & 15;
    int o = tid >> 4;
    float s = blin[o];
#pragma unroll 8
    for (int k = 0; k < HID; ++k) s += h2f[r * 64 + k] * Wlin[o * HID + k];
    out[(b0 + r) * OUTP + o] = s;
  }
}

extern "C" void kernel_launch(void* const* d_in, const int* in_sizes, int n_in,
                              void* d_out, int out_size, void* d_ws, size_t ws_size,
                              hipStream_t stream) {
  (void)in_sizes; (void)n_in; (void)out_size; (void)d_ws; (void)ws_size;
  const float* x    = (const float*)d_in[0];
  const float* Wih0 = (const float*)d_in[1];
  const float* Whh0 = (const float*)d_in[2];
  const float* bih0 = (const float*)d_in[3];
  const float* bhh0 = (const float*)d_in[4];
  const float* Wih1 = (const float*)d_in[5];
  const float* Whh1 = (const float*)d_in[6];
  const float* bih1 = (const float*)d_in[7];
  const float* bhh1 = (const float*)d_in[8];
  const float* Wlin = (const float*)d_in[9];
  const float* blin = (const float*)d_in[10];

  dim3 grid(BATCH / 16);   // 128 workgroups, one 16-row batch tile each
  dim3 block(128);         // 4 waves: one per 16-hidden-unit slice
  lstm2_fused_kernel<<<grid, block, 0, stream>>>(
      x, Wih0, Whh0, bih0, bhh0, Wih1, Whh1, bih1, bhh1, Wlin, blin,
      (float*)d_out);
}
